// Embedding_22436909154638
// MI455X (gfx1250) — matrix operand order repaired and checked by reading, not compiled, not run on hardware
//
#include <hip/hip_runtime.h>
#include <hip/hip_bf16.h>

typedef unsigned short u16;
typedef __attribute__((ext_vector_type(16))) __bf16 v16bf;
typedef __attribute__((ext_vector_type(8)))  float  v8f;
typedef __attribute__((ext_vector_type(16))) unsigned short ushort16;

#define NPTS   2048
#define KNN_K  32
#define BATCH  8
#define NPOS   (BATCH * NPTS * KNN_K)   /* 524288 positions per conv layer */
#define NEG_BIG (-3.402823466e38f)

__device__ __forceinline__ u16 f2bf(float f) {
    unsigned u = __builtin_bit_cast(unsigned, f);
    u += 0x7fffu + ((u >> 16) & 1u);          // round-to-nearest-even
    return (u16)(u >> 16);
}
__device__ __forceinline__ float lrelu(float y) { return y > 0.f ? y : 0.2f * y; }

// One conflict-free lane exchange: ds_swizzle group-of-32 bit-mode.
// offset = (xor_mask<<10) | (or_mask<<5) | and_mask  (and=0x1f keeps lane id)
template <int XORMASK>
__device__ __forceinline__ float swz_xor(float v) {
    int r = __builtin_amdgcn_ds_swizzle(__builtin_bit_cast(int, v),
                                        (XORMASK << 10) | 0x1f);
    return __builtin_bit_cast(float, r);
}

// Butterfly-sum across each 16-lane half (wave32): after this, every lane in a
// half holds the half's total.  Collapses the 16 lanes that share one output
// channel before touching LDS atomics (16x fewer ds_add_f32 conflicts), using
// single ds_swizzle_b32 ops instead of guarded ds_bpermute sequences.
__device__ __forceinline__ float half16_sum(float v) {
    v += swz_xor<1>(v);
    v += swz_xor<2>(v);
    v += swz_xor<4>(v);
    v += swz_xor<8>(v);
    return v;
}

// ---------------------------------------------------------------------------
// zero BN stat accumulators (4 layers x [sum128|sumsq128])
// ---------------------------------------------------------------------------
__global__ void zero_stats_kernel(float* stats) {
    for (int i = threadIdx.x; i < 1024; i += 256) stats[i] = 0.f;
}

// ---------------------------------------------------------------------------
// Pre-swizzle conv weights into WMMA bf16 A fragments, contiguous k order (same as B loads):
// lane l: M = mt*16 + (l%16); element e: v=e/2,o=e%2,
//   Klocal = e + (l>=16 ? 16 : 0)
// Fragment file: frag[((mt*4+kc)*32+lane)*16 + e]  (contiguous 32B per lane)
// ---------------------------------------------------------------------------
__device__ void make_frags(const float* __restrict__ W, u16* __restrict__ F,
                           int MT, int nthreads, int tid) {
    int total = MT * 4 * 32 * 16;
    for (int t = tid; t < total; t += nthreads) {
        int e    = t & 15;
        int lane = (t >> 4) & 31;
        int kc   = (t >> 9) & 3;
        int mt   = t >> 11;
        int v = e >> 1, o = e & 1;
        int kl = (2 * v + o) + ((lane >= 16) ? 16 : 0);
        int M = mt * 16 + (lane & 15);
        int K = kc * 32 + kl;
        F[t] = f2bf(W[M * 128 + K]);
    }
}

__global__ void prep_weights_kernel(const float* w2, const float* w3, const float* w4,
                                    u16* f2, u16* f3, u16* f4) {
    int tid = blockIdx.x * 256 + threadIdx.x;
    make_frags(w2, f2, 4, 64 * 256, tid);
    make_frags(w3, f3, 8, 64 * 256, tid);
    make_frags(w4, f4, 4, 64 * 256, tid);
}

// ---------------------------------------------------------------------------
// KNN: wave-per-row. dist rows buffered in LDS, candidate tiles DMA'd into
// LDS with gfx1250 async-to-LDS (ASYNCcnt, no VGPR round trip), shared by all
// 4 waves; 32 rounds of wave-wide argmax + invalidate.
// src layout (B, C, N).  rows per block = 4 (all in same batch: N%4==0).
// ---------------------------------------------------------------------------
template <int C>
__global__ void __launch_bounds__(128) knn_kernel(const float* __restrict__ src,
                                                  int* __restrict__ idx_out) {
    __shared__ float dist[4][NPTS];
    __shared__ float ctr[4][C];
    __shared__ float tile[C][64];
    int t = threadIdx.x, w = t >> 5, lane = t & 31;
    int row = blockIdx.x * 4 + w;
    int b = row >> 11, i = row & (NPTS - 1);

    for (int c = lane; c < C; c += 32) ctr[w][c] = src[(b * C + c) * NPTS + i];

    const float* sb = src + (b * C) * NPTS;     // block-uniform batch base (SGPR)
    for (int j0 = 0; j0 < NPTS; j0 += 64) {
        __syncthreads();
        for (int u = t; u < C * 64; u += 128) {
            int c = u >> 6, jj = u & 63;
            unsigned goff = (unsigned)((c * NPTS + j0 + jj) * 4);
            unsigned loff = (unsigned)(uintptr_t)(&tile[c][jj]);
            // Direct global->LDS DMA (CDNA5 async path, tracked by ASYNCcnt)
            asm volatile("global_load_async_to_lds_b32 %0, %1, %2"
                         :
                         : "v"(loff), "v"(goff), "s"(sb)
                         : "memory");
        }
        asm volatile("s_wait_asynccnt 0" ::: "memory");
        __syncthreads();
        for (int jj = lane; jj < 64; jj += 32) {
            float d = 0.f;
#pragma unroll
            for (int c = 0; c < C; c++) {
                float df = ctr[w][c] - tile[c][jj];
                d += df * df;
            }
            dist[w][j0 + jj] = -d;   // neg squared distance == reference neg_dist
        }
    }
    __syncthreads();
    for (int k = 0; k < KNN_K; k++) {
        float v = NEG_BIG; int bi = 0;
        for (int j = lane; j < NPTS; j += 32) {
            float d = dist[w][j];
            if (d > v) { v = d; bi = j; }      // ascending j keeps lowest index on tie
        }
#pragma unroll
        for (int off = 16; off > 0; off >>= 1) {
            float ov = __shfl_down(v, off);
            int  obi = __shfl_down(bi, off);
            if (ov > v || (ov == v && obi < bi)) { v = ov; bi = obi; }
        }
        if (lane == 0) { idx_out[row * KNN_K + k] = bi; dist[w][bi] = NEG_BIG; }
        __syncthreads();
    }
}

// ---------------------------------------------------------------------------
// conv1 stats: y1 = W1(128x6) @ feat1, accumulate per-channel sum / sumsq.
// 256 positions per block; thread owns one channel x 128 positions.
// ---------------------------------------------------------------------------
__global__ void __launch_bounds__(256) conv1_stats_kernel(const float* __restrict__ x,
                                                          const int* __restrict__ idx1,
                                                          const float* __restrict__ w1,
                                                          float* __restrict__ stats1) {
    __shared__ float feat[6][256];
    __shared__ float w1s[768];
    int t = threadIdx.x;
    int base = blockIdx.x * 256;
    for (int u = t; u < 768; u += 256) w1s[u] = w1[u];
    {
        int p = base + t;
        int b = p >> 16, rem = p & 65535, n = rem >> 5;
        int j = idx1[p];
#pragma unroll
        for (int c = 0; c < 3; c++) {
            float xj = x[(b * 3 + c) * NPTS + j];
            float xn = x[(b * 3 + c) * NPTS + n];
            feat[c][t] = xj - xn;
            feat[c + 3][t] = xn;
        }
    }
    __syncthreads();
    int c = t & 127, half = t >> 7;
    float sum = 0.f, sq = 0.f;
    for (int pl = half * 128; pl < half * 128 + 128; pl++) {
        float y = 0.f;
#pragma unroll
        for (int cc = 0; cc < 6; cc++) y += w1s[c * 6 + cc] * feat[cc][pl];
        sum += y; sq += y * y;
    }
    atomicAdd(&stats1[c], sum);
    atomicAdd(&stats1[128 + c], sq);
}

// ---------------------------------------------------------------------------
// BN finalize: scale = g * rsqrt(var+eps); shift = b - mean*scale
// ---------------------------------------------------------------------------
__global__ void bn_finalize_kernel(const float* stats, const float* gamma, const float* beta,
                                   float* bnp, int nch, float inv_count) {
    int t = threadIdx.x;
    if (t < nch) {
        float mean = stats[t] * inv_count;
        float var = stats[nch + t] * inv_count - mean * mean;
        float s = gamma[t] * rsqrtf(var + 1e-5f);
        bnp[t] = s;
        bnp[nch + t] = beta[t] - mean * s;
    }
}

// ---------------------------------------------------------------------------
// Stage-1 WMMA body: build a1 (bf16, pos-major [128pos][128ch]) in LDS,
// conv2 = w2(64x128) @ a1 via v_wmma_f32_16x16x32_bf16.
// MODE=0: accumulate BN2 stats.  MODE=1: apply BN2+lrelu, max over K -> h1/out.
// ---------------------------------------------------------------------------
template <int MODE>
__global__ void __launch_bounds__(256) s1_conv2_kernel(const float* __restrict__ x,
                                                       const int* __restrict__ idx1,
                                                       const float* __restrict__ w1,
                                                       const float* __restrict__ bnp1,
                                                       const float* __restrict__ bnp2,
                                                       const u16* __restrict__ w2f,
                                                       float* __restrict__ stats2,
                                                       float* __restrict__ h1,
                                                       float* __restrict__ out) {
    __shared__ float feat[6][128];
    __shared__ float w1s[768];
    __shared__ float bn1s[256];
    __shared__ float bn2s[128];
    __shared__ __align__(32) u16 a1[128 * 128];   // also aliased as f32 act[128][64]
    __shared__ float sst[128];
    int t = threadIdx.x;
    int base = blockIdx.x * 128;

    for (int u = t; u < 768; u += 256) w1s[u] = w1[u];
    if (t < 256) bn1s[t] = bnp1[t];
    if (MODE == 1 && t < 128) bn2s[t] = bnp2[t];
    if (t < 128) sst[t] = 0.f;
    if (t < 128) {
        int p = base + t;
        int b = p >> 16, rem = p & 65535, n = rem >> 5;
        int j = idx1[p];
#pragma unroll
        for (int c = 0; c < 3; c++) {
            float xj = x[(b * 3 + c) * NPTS + j];
            float xn = x[(b * 3 + c) * NPTS + n];
            feat[c][t] = xj - xn;
            feat[c + 3][t] = xn;
        }
    }
    __syncthreads();
    for (int u = t; u < 16384; u += 256) {
        int pl = u >> 7, c = u & 127;
        float y = 0.f;
#pragma unroll
        for (int cc = 0; cc < 6; cc++) y += w1s[c * 6 + cc] * feat[cc][pl];
        y = bn1s[c] * y + bn1s[128 + c];
        a1[pl * 128 + c] = f2bf(lrelu(y));
    }
    __syncthreads();

    int wave = t >> 5, lane = t & 31;
    int prow = wave * 16 + (lane & 15);
    int bsel = ((lane >> 4) & 1) * 16;
    int ohalf = (lane >= 16) ? 8 : 0;

    v16bf bfr[4];
#pragma unroll
    for (int kc = 0; kc < 4; kc++)
        bfr[kc] = __builtin_bit_cast(v16bf,
            *reinterpret_cast<const ushort16*>(&a1[prow * 128 + kc * 32 + bsel]));

    float* actf = reinterpret_cast<float*>(a1);   // [pos][64] f32, reused region
#pragma unroll
    for (int mt = 0; mt < 4; mt++) {
        v8f acc = {};
#pragma unroll
        for (int kc = 0; kc < 4; kc++) {
            v16bf av = __builtin_bit_cast(v16bf,
                *reinterpret_cast<const ushort16*>(w2f + ((mt * 4 + kc) * 32 + lane) * 16));
            acc = __builtin_amdgcn_wmma_f32_16x16x32_bf16(false, av, false, bfr[kc],
                                                          (short)0, acc, false, false);
        }
#pragma unroll
        for (int r = 0; r < 8; r++) {
            int o = mt * 16 + r + ohalf;
            float v = acc[r];
            if (MODE == 0) {
                float s  = half16_sum(v);
                float s2 = half16_sum(v * v);
                if ((lane & 15) == 0) {
                    atomicAdd(&sst[o], s);
                    atomicAdd(&sst[64 + o], s2);
                }
            } else {
                float y = bn2s[o] * v + bn2s[64 + o];
                actf[prow * 64 + o] = lrelu(y);
            }
        }
    }
    __syncthreads();
    if (MODE == 0) {
        if (t < 128) atomicAdd(&stats2[t], sst[t]);
    } else {
        int o = t & 63, n4 = t >> 6;        // 4 n-groups of 32 k each
        float m = NEG_BIG;
#pragma unroll
        for (int k = 0; k < KNN_K; k++) m = fmaxf(m, actf[(n4 * KNN_K + k) * 64 + o]);
        int b = base >> 16, rem = base & 65535;
        int n = (rem >> 5) + n4;
        h1[(b * 64 + o) * NPTS + n] = m;
        out[(b * 128 + o) * NPTS + n] = m;
    }
}

// ---------------------------------------------------------------------------
// Stage-2 conv3 stats: feat3 (128ch) gathered from h1, conv3 = w3(128x128)
// ---------------------------------------------------------------------------
__global__ void __launch_bounds__(256) s2_conv3_stats_kernel(const float* __restrict__ h1,
                                                             const int* __restrict__ idx2,
                                                             const u16* __restrict__ w3f,
                                                             float* __restrict__ stats3) {
    __shared__ __align__(32) u16 f3[128 * 128];
    __shared__ float sst[256];
    int t = threadIdx.x;
    int base = blockIdx.x * 128;
    if (t < 256) sst[t] = 0.f;
    for (int u = t; u < 16384; u += 256) {
        int pl = u >> 7, c = u & 127;
        int p = base + pl;
        int b = p >> 16, rem = p & 65535, n = rem >> 5;
        int j = idx2[p];
        float val = (c < 64) ? h1[(b * 64 + c) * NPTS + j] - h1[(b * 64 + c) * NPTS + n]
                             : h1[(b * 64 + (c - 64)) * NPTS + n];
        f3[pl * 128 + c] = f2bf(val);
    }
    __syncthreads();
    int wave = t >> 5, lane = t & 31;
    int prow = wave * 16 + (lane & 15);
    int bsel = ((lane >> 4) & 1) * 16;
    int ohalf = (lane >= 16) ? 8 : 0;
    v16bf bfr[4];
#pragma unroll
    for (int kc = 0; kc < 4; kc++)
        bfr[kc] = __builtin_bit_cast(v16bf,
            *reinterpret_cast<const ushort16*>(&f3[prow * 128 + kc * 32 + bsel]));
#pragma unroll
    for (int mt = 0; mt < 8; mt++) {
        v8f acc = {};
#pragma unroll
        for (int kc = 0; kc < 4; kc++) {
            v16bf av = __builtin_bit_cast(v16bf,
                *reinterpret_cast<const ushort16*>(w3f + ((mt * 4 + kc) * 32 + lane) * 16));
            acc = __builtin_amdgcn_wmma_f32_16x16x32_bf16(false, av, false, bfr[kc],
                                                          (short)0, acc, false, false);
        }
#pragma unroll
        for (int r = 0; r < 8; r++) {
            int o = mt * 16 + r + ohalf;
            float v = acc[r];
            float s  = half16_sum(v);
            float s2 = half16_sum(v * v);
            if ((lane & 15) == 0) {
                atomicAdd(&sst[o], s);
                atomicAdd(&sst[128 + o], s2);
            }
        }
    }
    __syncthreads();
    atomicAdd(&stats3[t], sst[t]);
}

// ---------------------------------------------------------------------------
// Stage-2 fused conv3 -> BN3+lrelu -> conv4.  64 positions/block, 8 waves:
// wave pairs split the M tiles of conv3 (o 0..63 / 64..127) and of conv4.
// MODE=0: BN4 stats.  MODE=1: BN4+lrelu + max over K -> out[:,64:,:]
// ---------------------------------------------------------------------------
template <int MODE>
__global__ void __launch_bounds__(256) s2_conv4_kernel(const float* __restrict__ h1,
                                                       const int* __restrict__ idx2,
                                                       const u16* __restrict__ w3f,
                                                       const u16* __restrict__ w4f,
                                                       const float* __restrict__ bnp3,
                                                       const float* __restrict__ bnp4,
                                                       float* __restrict__ stats4,
                                                       float* __restrict__ out) {
    __shared__ __align__(32) u16 f3[64 * 128];   // MODE=1: reused as f32 act4[64][64]
    __shared__ __align__(32) u16 a3[64 * 128];
    __shared__ float bn3s[256];
    __shared__ float bn4s[128];
    __shared__ float sst[128];
    int t = threadIdx.x;
    int base = blockIdx.x * 64;
    if (t < 256) bn3s[t] = bnp3[t];
    if (MODE == 1 && t < 128) bn4s[t] = bnp4[t];
    if (t < 128) sst[t] = 0.f;
    for (int u = t; u < 8192; u += 256) {
        int pl = u >> 7, c = u & 127;
        int p = base + pl;
        int b = p >> 16, rem = p & 65535, n = rem >> 5;
        int j = idx2[p];
        if ((u & 1023) == 0) __builtin_prefetch(&idx2[p + 16], 0, 1);
        float val = (c < 64) ? h1[(b * 64 + c) * NPTS + j] - h1[(b * 64 + c) * NPTS + n]
                             : h1[(b * 64 + (c - 64)) * NPTS + n];
        f3[pl * 128 + c] = f2bf(val);
    }
    __syncthreads();
    int wave = t >> 5, lane = t & 31;
    int prow = (wave & 3) * 16 + (lane & 15);
    int bsel = ((lane >> 4) & 1) * 16;
    int ohalf = (lane >= 16) ? 8 : 0;

    {   // conv3 + BN3 + lrelu -> a3 (bf16)
        v16bf bfr[4];
#pragma unroll
        for (int kc = 0; kc < 4; kc++)
            bfr[kc] = __builtin_bit_cast(v16bf,
                *reinterpret_cast<const ushort16*>(&f3[prow * 128 + kc * 32 + bsel]));
        int mt0 = (wave >> 2) * 4;
#pragma unroll
        for (int mi = 0; mi < 4; mi++) {
            int mt = mt0 + mi;
            v8f acc = {};
#pragma unroll
            for (int kc = 0; kc < 4; kc++) {
                v16bf av = __builtin_bit_cast(v16bf,
                    *reinterpret_cast<const ushort16*>(w3f + ((mt * 4 + kc) * 32 + lane) * 16));
                acc = __builtin_amdgcn_wmma_f32_16x16x32_bf16(false, av, false, bfr[kc],
                                                              (short)0, acc, false, false);
            }
#pragma unroll
            for (int r = 0; r < 8; r++) {
                int o = mt * 16 + r + ohalf;
                float y = bn3s[o] * acc[r] + bn3s[128 + o];
                a3[prow * 128 + o] = f2bf(lrelu(y));
            }
        }
    }
    __syncthreads();
    float* act4 = reinterpret_cast<float*>(f3);
    {   // conv4
        v16bf bfr[4];
#pragma unroll
        for (int kc = 0; kc < 4; kc++)
            bfr[kc] = __builtin_bit_cast(v16bf,
                *reinterpret_cast<const ushort16*>(&a3[prow * 128 + kc * 32 + bsel]));
        int mt0 = (wave >> 2) * 2;
#pragma unroll
        for (int mi = 0; mi < 2; mi++) {
            int mt = mt0 + mi;
            v8f acc = {};
#pragma unroll
            for (int kc = 0; kc < 4; kc++) {
                v16bf av = __builtin_bit_cast(v16bf,
                    *reinterpret_cast<const ushort16*>(w4f + ((mt * 4 + kc) * 32 + lane) * 16));
                acc = __builtin_amdgcn_wmma_f32_16x16x32_bf16(false, av, false, bfr[kc],
                                                              (short)0, acc, false, false);
            }
#pragma unroll
            for (int r = 0; r < 8; r++) {
                int o = mt * 16 + r + ohalf;
                float v = acc[r];
                if (MODE == 0) {
                    float s  = half16_sum(v);
                    float s2 = half16_sum(v * v);
                    if ((lane & 15) == 0) {
                        atomicAdd(&sst[o], s);
                        atomicAdd(&sst[64 + o], s2);
                    }
                } else {
                    float y = bn4s[o] * v + bn4s[64 + o];
                    act4[prow * 64 + o] = lrelu(y);
                }
            }
        }
    }
    __syncthreads();
    if (MODE == 0) {
        if (t < 128) atomicAdd(&stats4[t], sst[t]);
    } else if (t < 128) {
        int o = t & 63, n2 = t >> 6;   // 2 n-groups of 32 k
        float m = NEG_BIG;
#pragma unroll
        for (int k = 0; k < KNN_K; k++) m = fmaxf(m, act4[(n2 * KNN_K + k) * 64 + o]);
        int b = base >> 16, rem = base & 65535;
        int n = (rem >> 5) + n2;
        out[(b * 128 + 64 + o) * NPTS + n] = m;
    }
}

// ---------------------------------------------------------------------------
extern "C" void kernel_launch(void* const* d_in, const int* in_sizes, int n_in,
                              void* d_out, int out_size, void* d_ws, size_t ws_size,
                              hipStream_t stream) {
    const float* x  = (const float*)d_in[0];
    const float* w1 = (const float*)d_in[1];
    const float* g1 = (const float*)d_in[2];
    const float* b1 = (const float*)d_in[3];
    const float* w2 = (const float*)d_in[4];
    const float* g2 = (const float*)d_in[5];
    const float* b2 = (const float*)d_in[6];
    const float* w3 = (const float*)d_in[7];
    const float* g3 = (const float*)d_in[8];
    const float* b3 = (const float*)d_in[9];
    const float* w4 = (const float*)d_in[10];
    const float* g4 = (const float*)d_in[11];
    const float* b4 = (const float*)d_in[12];
    float* out = (float*)d_out;

    float* wsf   = (float*)d_ws;
    float* h1    = wsf;                    // 1,048,576 f32 (4 MB)
    float* stats = wsf + 1048576;          // 1024 f32: 4 layers x [sum|sumsq]
    float* bnp   = stats + 1024;           // 1024 f32: 4 layers x [scale|shift]
    int*   idx1  = (int*)(bnp + 1024);     // 524288 int
    int*   idx2  = idx1 + NPOS;            // 524288 int
    u16*   wf2   = (u16*)(idx2 + NPOS);    // 8192 bf16 (32B-aligned offset)
    u16*   wf3   = wf2 + 8192;             // 16384 bf16
    u16*   wf4   = wf3 + 16384;            // 8192 bf16

    const float inv_count = 1.0f / (float)NPOS;

    zero_stats_kernel<<<1, 256, 0, stream>>>(stats);
    prep_weights_kernel<<<64, 256, 0, stream>>>(w2, w3, w4, wf2, wf3, wf4);

    // ---- stage 1 ----
    knn_kernel<3><<<BATCH * NPTS / 4, 128, 0, stream>>>(x, idx1);
    conv1_stats_kernel<<<NPOS / 256, 256, 0, stream>>>(x, idx1, w1, stats);
    bn_finalize_kernel<<<1, 128, 0, stream>>>(stats, g1, b1, bnp, 128, inv_count);
    s1_conv2_kernel<0><<<NPOS / 128, 256, 0, stream>>>(x, idx1, w1, bnp, bnp + 256,
                                                       wf2, stats + 256, h1, out);
    bn_finalize_kernel<<<1, 128, 0, stream>>>(stats + 256, g2, b2, bnp + 256, 64, inv_count);
    s1_conv2_kernel<1><<<NPOS / 128, 256, 0, stream>>>(x, idx1, w1, bnp, bnp + 256,
                                                       wf2, stats + 256, h1, out);

    // ---- stage 2 ----
    knn_kernel<64><<<BATCH * NPTS / 4, 128, 0, stream>>>(h1, idx2);
    s2_conv3_stats_kernel<<<NPOS / 128, 256, 0, stream>>>(h1, idx2, wf3, stats + 512);
    bn_finalize_kernel<<<1, 128, 0, stream>>>(stats + 512, g3, b3, bnp + 512, 128, inv_count);
    s2_conv4_kernel<0><<<NPOS / 64, 256, 0, stream>>>(h1, idx2, wf3, wf4, bnp + 512,
                                                      bnp + 768, stats + 768, out);
    bn_finalize_kernel<<<1, 128, 0, stream>>>(stats + 768, g4, b4, bnp + 768, 64, inv_count);
    s2_conv4_kernel<1><<<NPOS / 64, 256, 0, stream>>>(h1, idx2, wf3, wf4, bnp + 512,
                                                      bnp + 768, stats + 768, out);

    (void)in_sizes; (void)n_in; (void)out_size; (void)ws_size;
}
